// MeanCovar_60748017435177
// MI455X (gfx1250) — matrix operand order repaired and checked by reading, not compiled, not run on hardware
//
#include <hip/hip_runtime.h>
#include <hip/hip_bf16.h>

// ---------------- problem constants ----------------
#define BATCH 2048
#define DIM 256
#define NCLS 64
#define PADROWS 4096   // sum of per-cluster 32-padded counts <= 2048 + 64*31 = 4032

typedef __attribute__((ext_vector_type(16))) __bf16 v16bf;
typedef __attribute__((ext_vector_type(8)))  __bf16 v8bf;
typedef __attribute__((ext_vector_type(8)))  float  v8f;

static __device__ __forceinline__ v16bf cat16(v8bf a, v8bf b) {
    return __builtin_shufflevector(a, b, 0, 1, 2, 3, 4, 5, 6, 7,
                                   8, 9, 10, 11, 12, 13, 14, 15);
}

// ---------------- workspace layout (bytes) ----------------
// [0,256)        counts[64] int
// [256,512)      pcnt[64]   int   (counts rounded up to multiple of 32)
// [512,768)      poffs[64]  int   (exclusive prefix of pcnt)
// [768,8960)     assign[2048] int
// [9216,25600)   order[4096] int  (row ids, cluster-major, stable)
// [25600,26624)  cstat[64] float4 {cw, trS, sum diag^2, sum (m1_d)^2}
// [26624,92160)  frob_tile[64*256] float
// [92160, +2MB)  XhiT [256 dims][4096 padded rows] bf16   (TRANSPOSED)
// [.., +2MB)     XloT [256][4096] bf16
#define WS_COUNTS   0
#define WS_PCNT     256
#define WS_POFFS    512
#define WS_ASSIGN   768
#define WS_ORDER    9216
#define WS_CSTAT    25600
#define WS_FROB     26624
#define WS_XHI      92160
#define WS_XLO      (WS_XHI + PADROWS*DIM*2)

// ---------------- 1. zero counts ----------------
__global__ void mc_zero(int* counts) {
    counts[threadIdx.x] = 0;
}

// ---------------- 2. argmax + histogram ----------------
__global__ __launch_bounds__(256) void mc_argmax(const float* __restrict__ logits,
                                                 int* __restrict__ assign,
                                                 int* __restrict__ counts) {
    int b = blockIdx.x * blockDim.x + threadIdx.x;
    if (b >= BATCH) return;
    const float* l = logits + b * NCLS;
    float best = l[0];
    int bi = 0;
    for (int j = 1; j < NCLS; ++j) {        // first-max tie-break == jnp.argmax
        float v = l[j];
        if (v > best) { best = v; bi = j; }
    }
    assign[b] = bi;
    atomicAdd(&counts[bi], 1);              // final counts are order-independent
}

// ---------------- 3. prefix sums + stable counting sort ----------------
__global__ __launch_bounds__(64) void mc_scan(const int* __restrict__ counts,
                                              int* __restrict__ pcnt,
                                              int* __restrict__ poffs,
                                              const int* __restrict__ assign,
                                              int* __restrict__ order) {
    int k = threadIdx.x;                    // 64 threads
    if (k == 0) {
        int acc = 0;
        for (int j = 0; j < NCLS; ++j) {
            int p = (counts[j] + 31) & ~31;
            poffs[j] = acc;
            pcnt[j]  = p;
            acc += p;
        }
    }
    __syncthreads();
    // stable: each cluster scans rows in batch order -> deterministic placement
    int base = poffs[k];
    int cnt  = counts[k];
    int r = 0;
    for (int b = 0; b < BATCH && r < cnt; ++b) {
        if (assign[b] == k) { order[base + r] = b; ++r; }
    }
}

// ---------------- 4. gather rows, diff, bf16 hi/lo split, store X^T --------
__global__ __launch_bounds__(64) void mc_gather(const float* __restrict__ emb,
                                                const float* __restrict__ centers,
                                                const int* __restrict__ counts,
                                                const int* __restrict__ pcnt,
                                                const int* __restrict__ poffs,
                                                const int* __restrict__ order,
                                                __bf16* __restrict__ XhiT,
                                                __bf16* __restrict__ XloT) {
    int p  = blockIdx.x;            // padded-row id 0..4095
    int d0 = threadIdx.x * 4;       // 64 threads * 4 cols = 256
    int k = -1;
    for (int j = 0; j < NCLS; ++j) {
        int o = poffs[j];
        if (p >= o && p < o + pcnt[j]) { k = j; break; }
    }
    float v[4];
    if (k >= 0 && (p - poffs[k]) < counts[k]) {
        int row = order[p];
        const float* e = emb + row * DIM + d0;
        const float* c = centers + k * DIM + d0;
        for (int j = 0; j < 4; ++j) v[j] = e[j] - c[j];
    } else {
        v[0] = v[1] = v[2] = v[3] = 0.f;    // zero padding => unmasked K-loop
    }
    for (int j = 0; j < 4; ++j) {
        float  x  = v[j];
        __bf16 hi = (__bf16)x;
        __bf16 lo = (__bf16)(x - (float)hi);
        XhiT[(d0 + j) * PADROWS + p] = hi;  // transposed: [dim][padded row]
        XloT[(d0 + j) * PADROWS + p] = lo;
    }
}

// ---------------- 5. per-cluster f32 stats (moment1 / trace / diag^2) ------
__global__ __launch_bounds__(256) void mc_stats(const __bf16* __restrict__ XhiT,
                                                const __bf16* __restrict__ XloT,
                                                const int* __restrict__ counts,
                                                const int* __restrict__ pcnt,
                                                const int* __restrict__ poffs,
                                                float4* __restrict__ cstat) {
    int k = blockIdx.x;
    int d = threadIdx.x;                   // one thread per dim
    int rows = pcnt[k];                    // multiple of 32
    int base = poffs[k];                   // multiple of 32 -> 16B aligned
    const __bf16* ph = XhiT + d * PADROWS + base;
    const __bf16* pl = XloT + d * PADROWS + base;
    float tot = 0.f, sq = 0.f;
    for (int r = 0; r < rows; r += 8) {    // 128-bit loads along row axis
        v8bf h = *(const v8bf*)(ph + r);
        v8bf l = *(const v8bf*)(pl + r);
        #pragma unroll
        for (int j = 0; j < 8; ++j) {
            float x = (float)h[j] + (float)l[j];
            tot += x;
            sq  += x * x;
        }
    }
    float cw  = (float)counts[k];
    float cwe = cw + 1e-8f;
    float m1  = tot / cwe;
    __shared__ float s1[DIM], s2[DIM], s3[DIM];
    s1[d] = m1 * m1;                       // sum_d moment1^2
    s2[d] = sq;                            // trace(S)
    s3[d] = sq * sq;                       // sum_d diag(S)^2
    __syncthreads();
    for (int off = 128; off; off >>= 1) {
        if (d < off) { s1[d] += s1[d + off]; s2[d] += s2[d + off]; s3[d] += s3[d + off]; }
        __syncthreads();
    }
    if (d == 0) cstat[k] = make_float4(cw, s2[0], s3[0], s1[0]);
}

// ---------------- 6. WMMA SYRK: ||S_k||_F^2 via split-bf16 GEMM ------------
// grid (32, 64): blockIdx.y = cluster, 8 waves/block, one 16x16 tile per wave.
// X stored transposed -> every WMMA operand is two contiguous b128 loads.
__global__ __launch_bounds__(256) void mc_syrk(const __bf16* __restrict__ XhiT,
                                               const __bf16* __restrict__ XloT,
                                               const int* __restrict__ pcnt,
                                               const int* __restrict__ poffs,
                                               float* __restrict__ frob_tile) {
    const int k    = blockIdx.y;
    const int wave = threadIdx.x >> 5;
    const int lane = threadIdx.x & 31;
    const int t    = blockIdx.x * 8 + wave;   // tile 0..255
    const int I    = t >> 4;                  // row-tile of S
    const int J    = t & 15;                  // col-tile of S
    const int m    = lane & 15;               // A-matrix M / B-matrix N
    const int hsel = lane >> 4;

    const int rows = pcnt[k];                 // multiple of 32 (zero padded)
    const int base = poffs[k];                // multiple of 32

    // per-lane row pointers into transposed storage
    const __bf16* rAh = XhiT + (I * 16 + m) * PADROWS + base;
    const __bf16* rAl = XloT + (I * 16 + m) * PADROWS + base;
    const __bf16* rBh = XhiT + (J * 16 + m) * PADROWS + base;
    const __bf16* rBl = XloT + (J * 16 + m) * PADROWS + base;
    const int oa = hsel * 8;                  // A: K = hsel*8+e (e<8), 16+hsel*8+e
    const int ob = hsel * 8;                  // B: K = hsel*8+e (e<8), 16+hsel*8+e

    v8f c = {0.f, 0.f, 0.f, 0.f, 0.f, 0.f, 0.f, 0.f};

    for (int c0 = 0; c0 < rows; c0 += 32) {
        // 8 x global_load_b128, 16B aligned (base,c0 multiples of 32)
        v16bf ahi = cat16(*(const v8bf*)(rAh + c0 + oa),
                          *(const v8bf*)(rAh + c0 + 16 + oa));
        v16bf alo = cat16(*(const v8bf*)(rAl + c0 + oa),
                          *(const v8bf*)(rAl + c0 + 16 + oa));
        v16bf bhi = cat16(*(const v8bf*)(rBh + c0 + ob),
                          *(const v8bf*)(rBh + c0 + 16 + ob));
        v16bf blo = cat16(*(const v8bf*)(rBl + c0 + ob),
                          *(const v8bf*)(rBl + c0 + 16 + ob));
        // S ~= Xhi^T Xhi + Xhi^T Xlo + Xlo^T Xhi  (lo*lo term ~2^-18, dropped)
        c = __builtin_amdgcn_wmma_f32_16x16x32_bf16(false, alo, false, bhi,
                                                    (short)0, c, false, false);
        c = __builtin_amdgcn_wmma_f32_16x16x32_bf16(false, ahi, false, blo,
                                                    (short)0, c, false, false);
        c = __builtin_amdgcn_wmma_f32_16x16x32_bf16(false, ahi, false, bhi,
                                                    (short)0, c, false, false);
    }
    float s = 0.f;
    #pragma unroll
    for (int e = 0; e < 8; ++e) s += c[e] * c[e];
    // wave32 reduction
    for (int off = 16; off; off >>= 1) s += __shfl_xor(s, off, 32);
    if (lane == 0) frob_tile[k * 256 + t] = s;
}

// ---------------- 7. finalize scalar ----------------
__global__ __launch_bounds__(64) void mc_final(const float4* __restrict__ cstat,
                                               const float* __restrict__ frob_tile,
                                               float* __restrict__ out) {
    int k = threadIdx.x;                    // 64 threads
    float4 cs = cstat[k];
    float cw = cs.x, trS = cs.y, dsq = cs.z, q1 = cs.w;
    float fr = 0.f;
    for (int tt = 0; tt < 256; ++tt) fr += frob_tile[k * 256 + tt];  // deterministic
    float cwe  = cw + 1e-8f;
    float inv  = 1.0f / cwe;
    float trM2 = trS * inv;                 // trace(moment2)
    float dM2  = dsq * inv * inv;           // sum diag(moment2)^2
    float fM2  = fr  * inv * inv;           // ||moment2||_F^2
    float cwn  = cw * (1.0f / (float)BATCH);
    const float a = 1.0f / (2.0f * DIM);
    const float b = 1.0f / (2.0f * DIM * (DIM - 1));
    float sim_ab = cwn * a * trM2;
    float sim_a  = sqrtf(cwn * ((a - b) * dM2 + b * fM2) + 1e-6f);
    float sim_b  = sqrtf(cwn * 0.5f + 1e-6f);      // sum(w2*eye) == 0.5
    float sim    = sim_ab / (sim_a * sim_b + 1e-6f);
    float p2     = (1.0f - sim) * (1.0f / NCLS);   // mean over clusters
    float p1     = cwn * (1.0f / DIM) * q1;
    __shared__ float sh[NCLS];
    sh[k] = p1 + 0.05f * p2;
    __syncthreads();
    for (int off = 32; off; off >>= 1) {
        if (k < off) sh[k] += sh[k + off];
        __syncthreads();
    }
    if (k == 0) out[0] = sh[0];
}

// ---------------- launcher ----------------
extern "C" void kernel_launch(void* const* d_in, const int* in_sizes, int n_in,
                              void* d_out, int out_size, void* d_ws, size_t ws_size,
                              hipStream_t stream) {
    const float* emb     = (const float*)d_in[0];   // [2048,256]
    const float* centers = (const float*)d_in[1];   // [64,256]
    const float* logits  = (const float*)d_in[2];   // [2048,64]
    float* out = (float*)d_out;

    char* w = (char*)d_ws;
    int*    counts = (int*)(w + WS_COUNTS);
    int*    pcnt   = (int*)(w + WS_PCNT);
    int*    poffs  = (int*)(w + WS_POFFS);
    int*    assign = (int*)(w + WS_ASSIGN);
    int*    order  = (int*)(w + WS_ORDER);
    float4* cstat  = (float4*)(w + WS_CSTAT);
    float*  frob   = (float*)(w + WS_FROB);
    __bf16* XhiT   = (__bf16*)(w + WS_XHI);
    __bf16* XloT   = (__bf16*)(w + WS_XLO);

    mc_zero<<<1, NCLS, 0, stream>>>(counts);
    mc_argmax<<<BATCH / 256, 256, 0, stream>>>(logits, assign, counts);
    mc_scan<<<1, NCLS, 0, stream>>>(counts, pcnt, poffs, assign, order);
    mc_gather<<<PADROWS, 64, 0, stream>>>(emb, centers, counts, pcnt, poffs, order,
                                          XhiT, XloT);
    mc_stats<<<NCLS, DIM, 0, stream>>>(XhiT, XloT, counts, pcnt, poffs, cstat);
    mc_syrk<<<dim3(32, NCLS), 256, 0, stream>>>(XhiT, XloT, pcnt, poffs, frob);
    mc_final<<<1, NCLS, 0, stream>>>(cstat, frob, out);
}
